// SaliencyModulatedAttention_88450556494631
// MI455X (gfx1250) — matrix operand order repaired and checked by reading, not compiled, not run on hardware
//
#include <hip/hip_runtime.h>
#include <hip/hip_bf16.h>
#include <math.h>

// ---------------------------------------------------------------------------
// SaliencyModulatedAttention for MI455X (gfx1250, wave32, WMMA).
//
// Roofline: 6 GEMMs of 16384x1024x1024 = 2.1e11 FLOP.  ~300MB of HBM traffic
// -> ~15-30us memory floor at 23.3 TB/s.  f16 WMMA (16x16x32, fp32
// accumulate) keeps matrix compute below the memory roof (fp32 WMMA 16x16x4
// would be ~8x slower per instruction and compute-bound), so activations and
// weights are converted to f16 once.  The tiny 16x16 cross-head attention
// (softmax) is kept fully fp32 using V_WMMA_F32_16X16X4_F32.
//
// GEMM tiling: one wave owns a 32x64 output strip (2 m-tiles x 4 n-tiles,
// 8 f32 accumulators).  A-fragment reuse x4 across n, B-fragment reuse x2
// across m -> B L2 traffic halved vs a 16x64 strip (~1.5GB total L2 reads).
// ---------------------------------------------------------------------------

typedef __attribute__((ext_vector_type(2)))  float    v2f;
typedef __attribute__((ext_vector_type(4)))  float    v4f;
typedef __attribute__((ext_vector_type(8)))  float    v8f;
typedef __attribute__((ext_vector_type(8)))  _Float16 v8h;
typedef __attribute__((ext_vector_type(16))) _Float16 v16h;

#define DM    1024          // d_model
#define MTOT  16384         // B*S
#define SEQ   4096

static __device__ __forceinline__ v16h cat8(v8h a, v8h b) {
  return __builtin_shufflevector(a, b, 0,1,2,3,4,5,6,7,8,9,10,11,12,13,14,15);
}

static __device__ __forceinline__ v8f wmma_f16(v16h a, v16h b, v8f c) {
  return __builtin_amdgcn_wmma_f32_16x16x32_f16(false, a, false, b, (short)0, c, false, false);
}

static __device__ __forceinline__ v8f wmma_f32k4(v2f a, v2f b, v8f c) {
  return __builtin_amdgcn_wmma_f32_16x16x4_f32(false, a, false, b, (short)0, c, false, false);
}

// ---------------------------------------------------------------------------
// fp32 -> f16 conversion, 4 elements/thread (all sizes are multiples of 1024).
// ---------------------------------------------------------------------------
__global__ __launch_bounds__(256) void cvt_f32_f16(const float* __restrict__ src,
                                                   _Float16* __restrict__ dst, int n) {
  int i = (blockIdx.x * 256 + threadIdx.x) * 4;
  if (i + 3 < n) {
    v4f s = *(const v4f*)(src + i);
    dst[i + 0] = (_Float16)s.x;
    dst[i + 1] = (_Float16)s.y;
    dst[i + 2] = (_Float16)s.z;
    dst[i + 3] = (_Float16)s.w;
  }
}

// ---------------------------------------------------------------------------
// Out[M=16384, N=1024] = A@W (+ A2@W2) + bias (+ bias2), f16 in / f32 out.
// One wave computes a 32x64 strip: 2 m-tiles x 4 n-tiles, 8 accumulators.
// A fragment (16x32 f16): lane m = lane%16, K chunks [8*hi,8*hi+8) and
// [16+8*hi,16+8*hi+8) -> two contiguous b128 loads from row-major A.
// B fragment (32x16 f16): n = lane%16, K of element s same as A's -> 16 halves
// down column n of row-major W -> 16 strided loads, shared by both m-tiles.
// C/D (16x16 f32): vgpr r = rows r / r+8 across the two 16-lane halves.
// ---------------------------------------------------------------------------
__global__ __launch_bounds__(256) void gemm_f16(
    const _Float16* __restrict__ A,  const _Float16* __restrict__ W,
    const float*    __restrict__ bias,
    const _Float16* __restrict__ A2, const _Float16* __restrict__ W2,
    const float*    __restrict__ bias2,
    float* __restrict__ Out)
{
  const int lane = threadIdx.x & 31;
  const int lo   = lane & 15;
  const int hi   = lane >> 4;
  const int w    = blockIdx.x * 8 + (threadIdx.x >> 5);  // global wave id
  const int msup = w >> 4;                               // 512 super-tiles (32 rows)
  const int ncol = (w & 15) * 64;                        // 16 n-strips of 64
  const int m0   = msup * 32 + lo;                       // m-tile 0 row; tile 1 = +16

  v8f acc[2][4];
  #pragma unroll
  for (int j = 0; j < 4; ++j) {
    float bb = bias[ncol + j * 16 + lo];
    if (bias2) bb += bias2[ncol + j * 16 + lo];
    #pragma unroll
    for (int e = 0; e < 8; ++e) { acc[0][j][e] = bb; acc[1][j][e] = bb; }
  }

  {
    const _Float16* ap0 = A + (size_t)m0 * DM + 8 * hi;
    const _Float16* ap1 = ap0 + (size_t)16 * DM;
    for (int kb = 0; kb < DM; kb += 32) {
      __builtin_prefetch(ap0 + kb + 128, 0, 1);
      v16h a0 = cat8(*(const v8h*)(ap0 + kb), *(const v8h*)(ap0 + kb + 16));
      v16h a1 = cat8(*(const v8h*)(ap1 + kb), *(const v8h*)(ap1 + kb + 16));
      const _Float16* wp = W + (size_t)(kb + 8 * hi) * DM + ncol + lo;
      #pragma unroll
      for (int j = 0; j < 4; ++j) {
        v16h bfrag;
        #pragma unroll
        for (int s = 0; s < 16; ++s) bfrag[s] = wp[(size_t)(s + (s >= 8 ? 8 : 0)) * DM + j * 16];
        acc[0][j] = wmma_f16(a0, bfrag, acc[0][j]);
        acc[1][j] = wmma_f16(a1, bfrag, acc[1][j]);
      }
    }
  }
  if (A2) {
    const _Float16* ap0 = A2 + (size_t)m0 * DM + 8 * hi;
    const _Float16* ap1 = ap0 + (size_t)16 * DM;
    for (int kb = 0; kb < DM; kb += 32) {
      v16h a0 = cat8(*(const v8h*)(ap0 + kb), *(const v8h*)(ap0 + kb + 16));
      v16h a1 = cat8(*(const v8h*)(ap1 + kb), *(const v8h*)(ap1 + kb + 16));
      const _Float16* wp = W2 + (size_t)(kb + 8 * hi) * DM + ncol + lo;
      #pragma unroll
      for (int j = 0; j < 4; ++j) {
        v16h bfrag;
        #pragma unroll
        for (int s = 0; s < 16; ++s) bfrag[s] = wp[(size_t)(s + (s >= 8 ? 8 : 0)) * DM + j * 16];
        acc[0][j] = wmma_f16(a0, bfrag, acc[0][j]);
        acc[1][j] = wmma_f16(a1, bfrag, acc[1][j]);
      }
    }
  }

  #pragma unroll
  for (int t = 0; t < 2; ++t)
    #pragma unroll
    for (int j = 0; j < 4; ++j)
      #pragma unroll
      for (int r = 0; r < 8; ++r)
        Out[(size_t)(msup * 32 + t * 16 + r + 8 * hi) * DM + ncol + j * 16 + lo]
            = acc[t][j][r];
}

// ---------------------------------------------------------------------------
// Cross-head attention at each position: one wave per (b,s).
//  scores = q(16x64) @ k^T(64x16) / 8   via 16 steps of V_WMMA_F32_16X16X4_F32
//  softmax over columns (xor-shuffle within the 16-lane row halves)
//  ctx    = attn(16x16) @ v(16x64)      via f32 WMMA (attn staged in LDS to
//                                        reshape C-layout into an A-fragment)
// ctx is written as f16 with the torch transpose/reinterpret scramble fused
// into the address:  ctx2[b, n*256 + s/16, (s%16)*64 + d] = ctx[b,s,n,d]
// ---------------------------------------------------------------------------
__global__ __launch_bounds__(256) void attn_heads(
    const float* __restrict__ q, const float* __restrict__ k,
    const float* __restrict__ v, _Float16* __restrict__ ctx2)
{
  __shared__ float smem[8 * 256];        // one 16x16 attn tile per wave
  const int lane = threadIdx.x & 31;
  const int lo   = lane & 15;
  const int hi   = lane >> 4;
  const int wave = threadIdx.x >> 5;
  const int pos  = blockIdx.x * 8 + wave;
  const int b    = pos >> 12;            // / 4096
  const int s    = pos & 4095;

  const float* qp = q + (size_t)pos * DM;
  const float* kp = k + (size_t)pos * DM;
  const float* vp = v + (size_t)pos * DM;

  // scores: A lane -> (m=lo, K=t*4+2*hi+{0,1}); B lane -> (n=lo, same K)
  v8f sc = {};
  #pragma unroll
  for (int t = 0; t < 16; ++t) {
    const int off = lo * 64 + t * 4 + 2 * hi;
    v2f a = *(const v2f*)(qp + off);
    v2f bb = *(const v2f*)(kp + off);
    sc = wmma_f32k4(a, bb, sc);
  }

  // softmax over the 16 columns of each row (columns live across lane halves)
  float* my = smem + wave * 256;
  #pragma unroll
  for (int r = 0; r < 8; ++r) {
    float val = sc[r] * 0.125f;          // 1/sqrt(64)
    float mx = val;
    mx = fmaxf(mx, __shfl_xor(mx, 1, 32));
    mx = fmaxf(mx, __shfl_xor(mx, 2, 32));
    mx = fmaxf(mx, __shfl_xor(mx, 4, 32));
    mx = fmaxf(mx, __shfl_xor(mx, 8, 32));
    float e = __expf(val - mx);
    float sum = e;
    sum += __shfl_xor(sum, 1, 32);
    sum += __shfl_xor(sum, 2, 32);
    sum += __shfl_xor(sum, 4, 32);
    sum += __shfl_xor(sum, 8, 32);
    my[(r + 8 * hi) * 16 + lo] = e / sum;   // row-major 16x16 attn in LDS
  }

  // ctx = attn @ v : A from LDS (row = lo), B from v (strided rows of 64)
  v8f cacc[4] = {};
  #pragma unroll
  for (int t = 0; t < 4; ++t) {
    v2f a = *(const v2f*)(my + lo * 16 + t * 4 + 2 * hi);
    const int krow = t * 4 + 2 * hi;
    #pragma unroll
    for (int j = 0; j < 4; ++j) {
      v2f bb;
      bb.x = vp[(krow + 0) * 64 + j * 16 + lo];
      bb.y = vp[(krow + 1) * 64 + j * 16 + lo];
      cacc[j] = wmma_f32k4(a, bb, cacc[j]);
    }
  }

  // scatter with the fused transpose/reinterpret; element (j,r) is
  // head n = r + 8*hi, dim d = j*16 + lo
  const size_t base = (size_t)b * (SEQ * (size_t)DM)
                    + (size_t)(s >> 4) * DM + (size_t)(s & 15) * 64;
  #pragma unroll
  for (int j = 0; j < 4; ++j)
    #pragma unroll
    for (int r = 0; r < 8; ++r) {
      const int n = r + 8 * hi;
      ctx2[base + (size_t)n * (256 * DM) + j * 16 + lo] = (_Float16)cacc[j][r];
    }
}

// ---------------------------------------------------------------------------
extern "C" void kernel_launch(void* const* d_in, const int* in_sizes, int n_in,
                              void* d_out, int out_size, void* d_ws, size_t ws_size,
                              hipStream_t stream) {
  (void)in_sizes; (void)n_in; (void)out_size; (void)ws_size;

  const float* x   = (const float*)d_in[0];
  const float* sal = (const float*)d_in[1];
  const float* Wq  = (const float*)d_in[2];
  const float* bq  = (const float*)d_in[3];
  const float* Wk  = (const float*)d_in[4];
  const float* bk  = (const float*)d_in[5];
  const float* Wv  = (const float*)d_in[6];
  const float* bv  = (const float*)d_in[7];
  const float* Wsq = (const float*)d_in[8];
  const float* bsq = (const float*)d_in[9];
  const float* Wsk = (const float*)d_in[10];
  const float* bsk = (const float*)d_in[11];
  const float* Wo  = (const float*)d_in[12];
  const float* bo  = (const float*)d_in[13];

  // workspace layout (bytes): 3x64MB f32 q/k/v, 3x32MB f16 acts, 6x2MB f16 W
  const size_t ACT_F32 = (size_t)MTOT * DM * sizeof(float);     // 64 MB
  const size_t ACT_F16 = (size_t)MTOT * DM * sizeof(_Float16);  // 32 MB
  const size_t W_F16   = (size_t)DM * DM * sizeof(_Float16);    //  2 MB
  char* ws = (char*)d_ws;
  float*    qbuf = (float*)(ws);
  float*    kbuf = (float*)(ws + ACT_F32);
  float*    vbuf = (float*)(ws + 2 * ACT_F32);
  _Float16* Xh   = (_Float16*)(ws + 3 * ACT_F32);
  _Float16* Sh   = (_Float16*)(ws + 3 * ACT_F32 + ACT_F16);
  _Float16* Ch   = (_Float16*)(ws + 3 * ACT_F32 + 2 * ACT_F16);
  _Float16* Whq  = (_Float16*)(ws + 3 * ACT_F32 + 3 * ACT_F16);
  _Float16* Whk  = (_Float16*)((char*)Whq + W_F16);
  _Float16* Whv  = (_Float16*)((char*)Whq + 2 * W_F16);
  _Float16* Whsq = (_Float16*)((char*)Whq + 3 * W_F16);
  _Float16* Whsk = (_Float16*)((char*)Whq + 4 * W_F16);
  _Float16* Who  = (_Float16*)((char*)Whq + 5 * W_F16);

  const int nAct = MTOT * DM;        // 16,777,216
  const int nW   = DM * DM;          //  1,048,576
  cvt_f32_f16<<<nAct / 1024, 256, 0, stream>>>(x,   Xh,   nAct);
  cvt_f32_f16<<<nAct / 1024, 256, 0, stream>>>(sal, Sh,   nAct);
  cvt_f32_f16<<<nW   / 1024, 256, 0, stream>>>(Wq,  Whq,  nW);
  cvt_f32_f16<<<nW   / 1024, 256, 0, stream>>>(Wk,  Whk,  nW);
  cvt_f32_f16<<<nW   / 1024, 256, 0, stream>>>(Wv,  Whv,  nW);
  cvt_f32_f16<<<nW   / 1024, 256, 0, stream>>>(Wsq, Whsq, nW);
  cvt_f32_f16<<<nW   / 1024, 256, 0, stream>>>(Wsk, Whsk, nW);
  cvt_f32_f16<<<nW   / 1024, 256, 0, stream>>>(Wo,  Who,  nW);

  // 8192 wave-tiles (512 m-supertiles x 16 n-strips) / 8 waves per block
  const int gemmBlocks = (MTOT / 32) * (DM / 64) / 8;   // 1024
  gemm_f16<<<gemmBlocks, 256, 0, stream>>>(Xh, Whq, bq, Sh, Whsq, bsq, qbuf);
  gemm_f16<<<gemmBlocks, 256, 0, stream>>>(Xh, Whk, bk, Sh, Whsk, bsk, kbuf);
  gemm_f16<<<gemmBlocks, 256, 0, stream>>>(Xh, Whv, bv, nullptr, nullptr, nullptr, vbuf);

  attn_heads<<<MTOT / 8, 256, 0, stream>>>(qbuf, kbuf, vbuf, Ch);

  gemm_f16<<<gemmBlocks, 256, 0, stream>>>(Ch, Who, bo, nullptr, nullptr, nullptr,
                                           (float*)d_out);
}
